// MultiHeadAttention_69810398429421
// MI455X (gfx1250) — matrix operand order repaired and checked by reading, not compiled, not run on hardware
//
#include <hip/hip_runtime.h>
#include <cstdint>

// Problem constants (fixed by the reference)
#define BATCH 4
#define SEQL  2048
#define DMOD  1024
#define NHEAD 16
#define HDIM  64

typedef _Float16 v16h __attribute__((ext_vector_type(16)));
typedef _Float16 v8h  __attribute__((ext_vector_type(8)));
typedef float    v8f  __attribute__((ext_vector_type(8)));
typedef int      v4i  __attribute__((vector_size(16)));

union Frag16 { v16h v; v8h h[2]; };

__device__ __forceinline__ v8f wmma_f16(v16h a, v16h b, v8f c) {
  return __builtin_amdgcn_wmma_f32_16x16x32_f16(false, a, false, b, (short)0, c,
                                                false, false);
}

// ---- gfx1250 async global->LDS path (guarded; falls back to plain copy) ----
#if defined(__has_builtin)
#  if __has_builtin(__builtin_amdgcn_global_load_async_to_lds_b128) && \
      __has_builtin(__builtin_amdgcn_s_wait_asynccnt)
#    define HAVE_ASYNC 1
#  endif
#  if __has_builtin(__builtin_amdgcn_wmma_f16_16x16x32_f16)
#    define HAVE_WMMA_F16ACC 1
#  endif
#endif
#ifndef HAVE_ASYNC
#  define HAVE_ASYNC 0
#endif
#ifndef HAVE_WMMA_F16ACC
#  define HAVE_WMMA_F16ACC 0
#endif

#define AS1 __attribute__((address_space(1)))
#define AS3 __attribute__((address_space(3)))

__device__ __forceinline__ void async_copy16(const _Float16* g, _Float16* l) {
#if HAVE_ASYNC
  v4i* gp = (v4i*)g;
  v4i* lp = (v4i*)l;
  __builtin_amdgcn_global_load_async_to_lds_b128((AS1 v4i*)gp, (AS3 v4i*)lp, 0, 0);
#else
  *(uint4*)l = *(const uint4*)g;
#endif
}
__device__ __forceinline__ void async_join() {
#if HAVE_ASYNC
  __builtin_amdgcn_s_wait_asynccnt(0);
#endif
}

// ---------------------------------------------------------------------------
// Elementwise f32 -> f16 (vectorized): enables fully-async GEMM staging.
// ---------------------------------------------------------------------------
__global__ __launch_bounds__(256)
void cvt_f32_f16(const float* __restrict__ X, _Float16* __restrict__ Y)
{
  size_t i = ((size_t)blockIdx.x * 256 + threadIdx.x) * 8;
  float4 a = ((const float4*)(X + i))[0];
  float4 b = ((const float4*)(X + i))[1];
  union { uint4 u; _Float16 h[8]; } o;
  o.h[0] = (_Float16)a.x; o.h[1] = (_Float16)a.y;
  o.h[2] = (_Float16)a.z; o.h[3] = (_Float16)a.w;
  o.h[4] = (_Float16)b.x; o.h[5] = (_Float16)b.y;
  o.h[6] = (_Float16)b.z; o.h[7] = (_Float16)b.w;
  *(uint4*)(Y + i) = o.u;
}

// ---------------------------------------------------------------------------
// Weight prep: W f32 [k][n] -> Wt f16 [n][k]  (one-time).
// ---------------------------------------------------------------------------
__global__ __launch_bounds__(256)
void prep_weights(const float* __restrict__ W, _Float16* __restrict__ Wt)
{
  __shared__ float tile[32][33];
  const int n0 = blockIdx.x * 32, k0 = blockIdx.y * 32;
  const int tx = threadIdx.x & 31, ty = threadIdx.x >> 5;
#pragma unroll
  for (int i = 0; i < 4; ++i)
    tile[ty + 8 * i][tx] = W[(size_t)(k0 + ty + 8 * i) * 1024 + n0 + tx];
  __syncthreads();
#pragma unroll
  for (int i = 0; i < 4; ++i)
    Wt[(size_t)(n0 + ty + 8 * i) * 1024 + k0 + tx] = (_Float16)tile[tx][ty + 8 * i];
}

// ---------------------------------------------------------------------------
// GEMM: Y = X(8192x1024 f16) @ W + bias, Wt pre-transposed f16 [n][k].
// OUTMODE 0: Y f16 head-major [B][H][L][dh]            (Q,K projections)
// OUTMODE 1: Y f16 head-major transposed [B][H][dh][L] (V projection)
// OUTMODE 2: Y f32 row-major [B*L][1024]               (final projection)
// Block 256 thr (8 waves). Block tile 128x128; wave tile 32(M)x64(N).
// Double-buffered, fully-async LDS staging.
// ---------------------------------------------------------------------------
template<int OUTMODE>
__global__ __launch_bounds__(256)
void gemm1024(const _Float16* __restrict__ Xh, const _Float16* __restrict__ Wt,
              const float* __restrict__ bias, void* __restrict__ Yv)
{
  __shared__ __align__(16) _Float16 ldsA[2][128 * 32];
  __shared__ __align__(16) _Float16 ldsBt[2][128 * 32];

  const int tid  = threadIdx.x;
  const int lane = tid & 31;
  const int w    = tid >> 5;
  const int wm   = w >> 1;
  const int wn   = w & 1;
  const int m0   = blockIdx.x * 128;
  const int n0   = blockIdx.y * 128;
  const int nn   = lane & 15;
  const int half = lane >> 4;

  v8f acc[2][4] = {};

  const int ar = tid >> 1, ac0 = (tid & 1) * 16;  // 128 rows x 32: 16 f16/thread

  auto stage = [&](int buf, int k0) {
    const _Float16* gB = Wt + (size_t)(n0 + ar) * 1024 + k0 + ac0;
    _Float16* lB = ldsBt[buf] + ar * 32 + ac0;
    async_copy16(gB, lB);
    async_copy16(gB + 8, lB + 8);
    const _Float16* gA = Xh + (size_t)(m0 + ar) * 1024 + k0 + ac0;
    _Float16* lA = ldsA[buf] + ar * 32 + ac0;
    async_copy16(gA, lA);
    async_copy16(gA + 8, lA + 8);
  };

  stage(0, 0);
  async_join();
  __syncthreads();

  for (int k0 = 0; k0 < 1024; k0 += 32) {
    const int cur = (k0 >> 5) & 1;
    if (k0 + 32 < 1024) stage(cur ^ 1, k0 + 32);   // overlap copy with compute

    Frag16 a[2];
#pragma unroll
    for (int qt = 0; qt < 2; ++qt) {
      const v8h* ap = (const v8h*)(ldsA[cur] + (wm * 32 + qt * 16 + nn) * 32);
      a[qt].h[0] = ap[half];
      a[qt].h[1] = ap[2 + half];
    }
#pragma unroll
    for (int j = 0; j < 4; ++j) {
      Frag16 bfr;
      const v8h* bp = (const v8h*)(ldsBt[cur] + (wn * 64 + j * 16 + nn) * 32);
      bfr.h[0] = bp[half];
      bfr.h[1] = bp[2 + half];
#pragma unroll
      for (int qt = 0; qt < 2; ++qt)
        acc[qt][j] = wmma_f16(a[qt].v, bfr.v, acc[qt][j]);
    }

    async_join();     // next buffer landed (this wave's copies)
    __syncthreads();  // all waves done reading cur; safe to overwrite next iter
  }

  _Float16* Yh = (_Float16*)Yv;
  float*    Yf = (float*)Yv;
#pragma unroll
  for (int qt = 0; qt < 2; ++qt)
#pragma unroll
    for (int j = 0; j < 4; ++j)
#pragma unroll
      for (int r = 0; r < 8; ++r) {
        int M = m0 + wm * 32 + qt * 16 + r + 8 * half;
        int N = n0 + wn * 64 + j * 16 + nn;
        float val = acc[qt][j][r] + bias[N];
        if (OUTMODE == 2) {
          Yf[(size_t)M * 1024 + N] = val;
        } else {
          int b = M >> 11, l = M & 2047;
          int hh = N >> 6, dd = N & 63;
          if (OUTMODE == 0)
            Yh[(((size_t)b * NHEAD + hh) * SEQL + l) * HDIM + dd] = (_Float16)val;
          else  // V: store head transposed [B][H][dh][L]
            Yh[(((size_t)b * NHEAD + hh) * HDIM + dd) * SEQL + l] = (_Float16)val;
        }
      }
}

// ---------------------------------------------------------------------------
// Fused attention: O = (Q K^T) V streamed over 32-key blocks, double-buffered.
// Block 256 thr (8 waves); each wave owns 32 queries (block covers 256).
// Q/K in [B][H][L][64]; V pre-transposed [B][H][64][L]. Grid = 64*8 = 512.
// ---------------------------------------------------------------------------
__global__ __launch_bounds__(256)
void attn_kernel(const _Float16* __restrict__ Qw, const _Float16* __restrict__ Kw,
                 const _Float16* __restrict__ Vw, _Float16* __restrict__ Aout)
{
  __shared__ __align__(16) _Float16 Klds[2][32 * 64];  // [key][dh]
  __shared__ __align__(16) _Float16 Vt[2][64 * 32];    // [dh][key]
  __shared__ __align__(16) _Float16 Sld[8 * 16 * 32];  // per-wave S transpose

  const int tid  = threadIdx.x;
  const int lane = tid & 31;
  const int w    = tid >> 5;
  const int nn   = lane & 15;
  const int half = lane >> 4;

  const int qchunk = blockIdx.x & 7;
  const int bh     = blockIdx.x >> 3;
  const int b      = bh >> 4;
  const int hh     = bh & 15;
  const size_t headbase = (size_t)bh * SEQL * HDIM;

  const int qbase = qchunk * 256 + w * 32;

  Frag16 qA[2][2];
#pragma unroll
  for (int qt = 0; qt < 2; ++qt) {
    const v8h* qp = (const v8h*)(Qw + headbase + (size_t)(qbase + qt * 16 + nn) * HDIM);
#pragma unroll
    for (int s = 0; s < 2; ++s) {
      qA[qt][s].h[0] = qp[4 * s + half];
      qA[qt][s].h[1] = qp[4 * s + half + 2];
    }
  }

  v8f O[2][4] = {};

  const int k_r = tid >> 3, k_c = (tid & 7) * 8;  // K: 32 rows x 64
  const int v_r = tid >> 2, v_c = (tid & 3) * 8;  // V: 64 rows x 32

  auto stageKV = [&](int buf, int kb) {
    async_copy16(Kw + headbase + (size_t)(kb * 32 + k_r) * HDIM + k_c,
                 Klds[buf] + k_r * 64 + k_c);
    async_copy16(Vw + headbase + (size_t)v_r * SEQL + kb * 32 + v_c,
                 Vt[buf] + v_r * 32 + v_c);
  };

  stageKV(0, 0);
  async_join();
  __syncthreads();

  const int NKB = SEQL / 32;
  for (int kb = 0; kb < NKB; ++kb) {
    const int cur = kb & 1;
    if (kb + 1 < NKB) stageKV(cur ^ 1, kb + 1);  // overlap copy with compute
    if (kb + 3 < NKB) {
      __builtin_prefetch(Kw + headbase + (size_t)((kb + 3) * 32 + k_r) * HDIM + k_c, 0, 0);
      __builtin_prefetch(Vw + headbase + (size_t)v_r * SEQL + (kb + 3) * 32 + v_c, 0, 0);
    }

    Frag16 kf[2][2], vf[4];
#pragma unroll
    for (int j = 0; j < 2; ++j) {
      const v8h* kp = (const v8h*)(Klds[cur] + (j * 16 + nn) * 64);
#pragma unroll
      for (int t = 0; t < 2; ++t) {
        kf[j][t].h[0] = kp[4 * t + half];
        kf[j][t].h[1] = kp[4 * t + half + 2];
      }
    }
#pragma unroll
    for (int j = 0; j < 4; ++j) {
      const v8h* vp = (const v8h*)(Vt[cur] + (j * 16 + nn) * 32);
      vf[j].h[0] = vp[half];
      vf[j].h[1] = vp[2 + half];
    }

#pragma unroll
    for (int qt = 0; qt < 2; ++qt) {
      // S(16x32) = Q_qt(16x64) @ K^T ; f16 accumulate when available
      // (16-bit C/D layout: elem e -> M = e + 8*half, same indexing as f32 C)
#pragma unroll
      for (int j = 0; j < 2; ++j) {
#if HAVE_WMMA_F16ACC
        v8h s = {};
        s = __builtin_amdgcn_wmma_f16_16x16x32_f16(false, qA[qt][0].v, false,
                                                   kf[j][0].v, (short)0, s,
                                                   false, false);
        s = __builtin_amdgcn_wmma_f16_16x16x32_f16(false, qA[qt][1].v, false,
                                                   kf[j][1].v, (short)0, s,
                                                   false, false);
#pragma unroll
        for (int r = 0; r < 8; ++r)
          Sld[w * 512 + (r + 8 * half) * 32 + j * 16 + nn] = s[r];
#else
        v8f s = {};
        s = wmma_f16(qA[qt][0].v, kf[j][0].v, s);
        s = wmma_f16(qA[qt][1].v, kf[j][1].v, s);
#pragma unroll
        for (int r = 0; r < 8; ++r)
          Sld[w * 512 + (r + 8 * half) * 32 + j * 16 + nn] = (_Float16)s[r];
#endif
      }
      // Per-wave buffer; DS ops from one wave are processed in order.
      Frag16 aS;
      const v8h* sp = (const v8h*)(Sld + w * 512 + nn * 32);
      aS.h[0] = sp[half];
      aS.h[1] = sp[2 + half];
#pragma unroll
      for (int j = 0; j < 4; ++j)
        O[qt][j] = wmma_f16(aS.v, vf[j].v, O[qt][j]);
    }

    async_join();
    __syncthreads();
  }

  // Write head output into concatenated layout [B][L][H*64] (f16)
#pragma unroll
  for (int qt = 0; qt < 2; ++qt)
#pragma unroll
    for (int j = 0; j < 4; ++j)
#pragma unroll
      for (int r = 0; r < 8; ++r) {
        int l = qbase + qt * 16 + r + 8 * half;
        int dcol = hh * HDIM + j * 16 + nn;
        Aout[((size_t)b * SEQL + l) * DMOD + dcol] = (_Float16)O[qt][j][r];
      }
}

// ---------------------------------------------------------------------------
extern "C" void kernel_launch(void* const* d_in, const int* in_sizes, int n_in,
                              void* d_out, int out_size, void* d_ws, size_t ws_size,
                              hipStream_t stream)
{
  const float* q  = (const float*)d_in[0];
  const float* k  = (const float*)d_in[1];
  const float* v  = (const float*)d_in[2];
  const float* Wq = (const float*)d_in[3];
  const float* bq = (const float*)d_in[4];
  const float* Wk = (const float*)d_in[5];
  const float* bk = (const float*)d_in[6];
  const float* Wv = (const float*)d_in[7];
  const float* bv = (const float*)d_in[8];
  const float* Wc = (const float*)d_in[9];
  const float* bc = (const float*)d_in[10];

  const size_t NE = (size_t)BATCH * NHEAD * SEQL * HDIM;  // 8,388,608 f16
  const size_t NW = (size_t)DMOD * DMOD;                  // 1,048,576 f16
  _Float16* Qh  = (_Float16*)d_ws;
  _Float16* Kh  = Qh + NE;
  _Float16* Vh  = Kh + NE;   // head-transposed [B][H][dh][L]
  _Float16* Ah  = Vh + NE;   // attention out [B][L][D]
  _Float16* X1  = Ah + NE;   // reused f16 activation scratch
  _Float16* Wqt = X1 + NE;
  _Float16* Wkt = Wqt + NW;
  _Float16* Wvt = Wkt + NW;
  _Float16* Wct = Wvt + NW;

  dim3 blk(256);
  dim3 pgrid(32, 32);
  prep_weights<<<pgrid, blk, 0, stream>>>(Wq, Wqt);
  prep_weights<<<pgrid, blk, 0, stream>>>(Wk, Wkt);
  prep_weights<<<pgrid, blk, 0, stream>>>(Wv, Wvt);
  prep_weights<<<pgrid, blk, 0, stream>>>(Wc, Wct);

  const int cgrid = (int)(NE / (256 * 8));  // 4096
  dim3 ggrid(8192 / 128, 1024 / 128);       // (64, 8)

  // Stream-serialized: X1 is safely reused for q, k, v in turn.
  cvt_f32_f16<<<cgrid, blk, 0, stream>>>(q, X1);
  gemm1024<0><<<ggrid, blk, 0, stream>>>(X1, Wqt, bq, (void*)Qh);
  cvt_f32_f16<<<cgrid, blk, 0, stream>>>(k, X1);
  gemm1024<0><<<ggrid, blk, 0, stream>>>(X1, Wkt, bk, (void*)Kh);
  cvt_f32_f16<<<cgrid, blk, 0, stream>>>(v, X1);
  gemm1024<1><<<ggrid, blk, 0, stream>>>(X1, Wvt, bv, (void*)Vh);

  attn_kernel<<<dim3(BATCH * NHEAD * (SEQL / 256)), blk, 0, stream>>>(Qh, Kh, Vh, Ah);

  gemm1024<2><<<ggrid, blk, 0, stream>>>(Ah, Wct, bc, d_out);
}
